// DiffusionModel_88948772700534
// MI455X (gfx1250) — matrix operand order repaired and checked by reading, not compiled, not run on hardware
//
#include <hip/hip_runtime.h>

// ---------------------------------------------------------------- types
typedef __attribute__((ext_vector_type(16))) __bf16 v16bf;
typedef __attribute__((ext_vector_type(8)))  float  v8f;

#define B_   16
#define N_   96
#define NN_  (96 * 96)
#define R_   (16 * 96 * 96)
#define DN_  256
#define DE_  128
#define EPS_ 1e-5f

__device__ __forceinline__ float leaky_(float x) { return x > 0.f ? x : 0.01f * x; }

// ---------------------------------------------------------------- f32 -> bf16 copy (weights)
__global__ void k_f2bf(const float* __restrict__ src, __bf16* __restrict__ dst, int n) {
  int i = blockIdx.x * 256 + threadIdx.x;
  if (i < n) dst[i] = (__bf16)src[i];
}

// ---------------------------------------------------------------- WMMA GEMM
// D[rows,Nout] = epilogue( A[rows,K](bf16) @ W[Nout,K]^T(bf16) + bias )
// Block tile 128x64 (8 waves as 4x2), each wave a 32x32 tile via 2x2 WMMAs:
// A fragments shared across 2 col-tiles, B fragments across 2 row-tiles, and
// 4 independent accumulator chains to hide v_wmma latency.
template <int ACT, bool FILM, bool RES, bool OBF, bool OF32>
__global__ __launch_bounds__(256) void k_gemm(
    const __bf16* __restrict__ A, const __bf16* __restrict__ W,
    const float* __restrict__ bias, const float* __restrict__ mul,
    const float* __restrict__ add, const float* __restrict__ res,
    __bf16* __restrict__ obf, float* __restrict__ of32,
    int K, int Nout, int rowsPerBatch) {
  const int lane = threadIdx.x & 31;
  const int wave = threadIdx.x >> 5;                 // 0..7
  const int rowBase = blockIdx.x * 128 + (wave & 3) * 32;
  const int colBase = blockIdx.y * 64 + (wave >> 2) * 32;
  const int lh  = lane >> 4;                         // lane half
  const int l15 = lane & 15;

  // A fragment: row = l15; element e<8 -> K = lh*8+e ; e>=8 -> K = 16+lh*8+(e-8)
  const __bf16* pa0 = A + (size_t)(rowBase + l15) * K + lh * 8;
  const __bf16* pa1 = pa0 + (size_t)16 * K;
  // B fragment: col = l15; element e<8 -> K = lh*8+e ; e>=8 -> K = 16+lh*8+(e-8)
  const __bf16* pb0 = W + (size_t)(colBase + l15) * K + lh * 8;
  const __bf16* pb1 = pb0 + (size_t)16 * K;

  v8f acc00 = {}, acc01 = {}, acc10 = {}, acc11 = {};
  for (int kk = 0; kk < K; kk += 32) {
    union { v16bf v; uint4 u[2]; } fa0, fa1, fb0, fb1;
    fa0.u[0] = *(const uint4*)(pa0);
    fa0.u[1] = *(const uint4*)(pa0 + 16);
    fa1.u[0] = *(const uint4*)(pa1);
    fa1.u[1] = *(const uint4*)(pa1 + 16);
    fb0.u[0] = *(const uint4*)(pb0);
    fb0.u[1] = *(const uint4*)(pb0 + 16);
    fb1.u[0] = *(const uint4*)(pb1);
    fb1.u[1] = *(const uint4*)(pb1 + 16);
    // prefetch streaming A operand ~1KB (16 K-steps) ahead
    __builtin_prefetch(pa0 + 512, 0, 0);
    __builtin_prefetch(pa1 + 512, 0, 0);
    acc00 = __builtin_amdgcn_wmma_f32_16x16x32_bf16(false, fa0.v, false, fb0.v,
                                                    (short)0, acc00, false, false);
    acc01 = __builtin_amdgcn_wmma_f32_16x16x32_bf16(false, fa0.v, false, fb1.v,
                                                    (short)0, acc01, false, false);
    acc10 = __builtin_amdgcn_wmma_f32_16x16x32_bf16(false, fa1.v, false, fb0.v,
                                                    (short)0, acc10, false, false);
    acc11 = __builtin_amdgcn_wmma_f32_16x16x32_bf16(false, fa1.v, false, fb1.v,
                                                    (short)0, acc11, false, false);
    pa0 += 32; pa1 += 32; pb0 += 32; pb1 += 32;
  }

  const int bidx = FILM ? (rowBase / rowsPerBatch) : 0;  // 128 | 9216, tiles never cross batch
#pragma unroll
  for (int tj = 0; tj < 2; ++tj) {
    const int col = colBase + tj * 16 + l15;
    const float bv = bias[col];
    float fm = 1.f, fa = 0.f;
    if (FILM) {
      fm = mul[bidx * Nout + col];
      fa = add[bidx * Nout + col];
    }
#pragma unroll
    for (int ti = 0; ti < 2; ++ti) {
      v8f a = (ti == 0) ? (tj == 0 ? acc00 : acc01) : (tj == 0 ? acc10 : acc11);
#pragma unroll
      for (int v = 0; v < 8; ++v) {
        int row = rowBase + ti * 16 + v + lh * 8;
        float x = a[v] + bv;
        if (FILM) x = x * fm + fa;
        if (ACT == 1) x = leaky_(x);
        if (RES) x += res[(size_t)row * Nout + col];
        if (OBF)  obf[(size_t)row * Nout + col] = (__bf16)x;
        if (OF32) of32[(size_t)row * Nout + col] = x;
      }
    }
  }
}

// ---------------------------------------------------------------- time embedding + in_c MLP
__global__ void k_cond(const int* __restrict__ t, const float* __restrict__ w1,
                       const float* __restrict__ b1, const float* __restrict__ w2,
                       const float* __restrict__ b2, float* __restrict__ cond) {
  __shared__ float emb[256], hid[256];
  int b = blockIdx.x, tid = threadIdx.x;
  float tv = (float)t[b];
  int s = tid >> 1;
  float scale = __expf((float)(2 * s) * (-logf(10000.f) / 256.f));
  float arg = tv * scale;
  emb[tid] = (tid & 1) ? __cosf(arg) : __sinf(arg);
  __syncthreads();
  float a = b1[tid];
  for (int k = 0; k < 256; ++k) a += emb[k] * w1[tid * 256 + k];
  hid[tid] = leaky_(a);
  __syncthreads();
  a = b2[tid];
  for (int k = 0; k < 256; ++k) a += hid[k] * w2[tid * 256 + k];
  cond[b * 256 + tid] = leaky_(a);
}

// ---------------------------------------------------------------- FiLM mul/add vectors
__global__ void k_film(const float* __restrict__ cond, const float* __restrict__ wm,
                       const float* __restrict__ bm, const float* __restrict__ wa,
                       const float* __restrict__ ba, float* __restrict__ mul,
                       float* __restrict__ add) {
  __shared__ float c[256];
  int b = blockIdx.x, t = threadIdx.x;
  c[t] = cond[b * 256 + t];
  __syncthreads();
  float am = bm[t], aa = ba[t];
  for (int k = 0; k < 256; ++k) {
    am += c[k] * wm[t * 256 + k];
    aa += c[k] * wa[t * 256 + k];
  }
  mul[b * 256 + t] = am;
  add[b * 256 + t] = aa;
}

// ---------------------------------------------------------------- node input MLP (20->256->256)
__global__ void k_in_n(const float* __restrict__ x, const float* __restrict__ w1,
                       const float* __restrict__ b1, const float* __restrict__ w2,
                       const float* __restrict__ b2, float* __restrict__ out) {
  __shared__ float xr[20], hid[256];
  int r = blockIdx.x, t = threadIdx.x;
  if (t < 20) xr[t] = x[r * 20 + t];
  __syncthreads();
  float a = b1[t];
  for (int k = 0; k < 20; ++k) a += xr[k] * w1[t * 20 + k];
  hid[t] = leaky_(a);
  __syncthreads();
  a = b2[t];
  for (int k = 0; k < 256; ++k) a += hid[k] * w2[t * 256 + k];
  out[r * 256 + t] = leaky_(a);
}

// ---------------------------------------------------------------- edge input layer 1 (5->128, K tiny -> VALU)
__global__ void k_in_e1(const float* __restrict__ e, const float* __restrict__ w,
                        const float* __restrict__ bias, __bf16* __restrict__ out) {
  size_t idx = (size_t)blockIdx.x * 256 + threadIdx.x;
  if (idx >= (size_t)R_ * 128) return;
  size_t r = idx >> 7;
  int c = (int)(idx & 127);
  const float* er = e + r * 5;
  float a = bias[c];
  for (int k = 0; k < 5; ++k) a += er[k] * w[c * 5 + k];
  out[idx] = (__bf16)leaky_(a);
}

// ---------------------------------------------------------------- gf = [edges | nodes_j | nodes_i] (bf16)
__global__ void k_pack_gf(const float* __restrict__ edges, const float* __restrict__ nodes,
                          __bf16* __restrict__ gf) {
  int r = blockIdx.x;
  int b = r / NN_;
  int rem = r - b * NN_;
  int i = rem / N_;
  int j = rem - i * N_;
  const float* er = edges + (size_t)r * 128;
  const float* nj = nodes + (size_t)(b * N_ + j) * 256;
  const float* ni = nodes + (size_t)(b * N_ + i) * 256;
  __bf16* o = gf + (size_t)r * 640;
  for (int c = threadIdx.x; c < 640; c += 256) {
    float v = (c < 128) ? er[c] : (c < 384 ? nj[c - 128] : ni[c - 384]);
    o[c] = (__bf16)v;
  }
}

// ---------------------------------------------------------------- wn = sqrt(sum_d w^2 / 256) per head
__global__ void k_wn(const __bf16* __restrict__ w, float* __restrict__ wn) {
  int idx = blockIdx.x * 256 + threadIdx.x;  // r*8 + h
  if (idx >= R_ * 8) return;
  int r = idx >> 3, h = idx & 7;
  const __bf16* p = w + (size_t)r * 256 + h * 32;
  float s = 0.f;
  for (int d = 0; d < 32; ++d) {
    float v = (float)p[d];
    s += v * v;
  }
  wn[idx] = sqrtf(s * (1.f / 256.f));
}

// ---------------------------------------------------------------- softmax over j + weighted sum of v
__global__ void k_attn(const float* __restrict__ wn, const __bf16* __restrict__ v,
                       float* __restrict__ out) {
  int bi = blockIdx.x;          // b*96 + i
  int t = threadIdx.x;
  int h = t >> 5, d = t & 31;
  size_t base = ((size_t)bi * 96) * 8 + h;          // wn[(b,i,j,h)]
  float mx = -1e30f;
  for (int j = 0; j < 96; ++j) mx = fmaxf(mx, wn[base + (size_t)j * 8]);
  float sum = 0.f;
  for (int j = 0; j < 96; ++j) sum += __expf(wn[base + (size_t)j * 8] - mx);
  float inv = 1.f / sum;
  size_t vbase = ((size_t)bi * 96) * 256 + h * 32 + d;
  float acc = 0.f;
  for (int j = 0; j < 96; ++j) {
    float a = __expf(wn[base + (size_t)j * 8] - mx) * inv;
    acc += a * (float)v[vbase + (size_t)j * 256];
  }
  out[(size_t)bi * 256 + t] = acc;
}

// ---------------------------------------------------------------- 256->256 linear + leaky (rows=1536)
__global__ void k_lin256(const float* __restrict__ x, const float* __restrict__ w,
                         const float* __restrict__ bias, float* __restrict__ out) {
  __shared__ float xr[256];
  int r = blockIdx.x, t = threadIdx.x;
  xr[t] = x[r * 256 + t];
  __syncthreads();
  float a = bias[t];
  for (int k = 0; k < 256; ++k) a += xr[k] * w[t * 256 + k];
  out[r * 256 + t] = leaky_(a);
}

// ---------------------------------------------------------------- 256->256->256 MLP, both leaky
__global__ void k_mlp256(const float* __restrict__ x, const float* __restrict__ w1,
                         const float* __restrict__ b1, const float* __restrict__ w2,
                         const float* __restrict__ b2, float* __restrict__ out) {
  __shared__ float xr[256], hid[256];
  int r = blockIdx.x, t = threadIdx.x;
  xr[t] = x[r * 256 + t];
  __syncthreads();
  float a = b1[t];
  for (int k = 0; k < 256; ++k) a += xr[k] * w1[t * 256 + k];
  hid[t] = leaky_(a);
  __syncthreads();
  a = b2[t];
  for (int k = 0; k < 256; ++k) a += hid[k] * w2[t * 256 + k];
  out[r * 256 + t] = leaky_(a);
}

// ---------------------------------------------------------------- node residual + InstanceNorm over i
__global__ void k_node_addnorm(const float* __restrict__ x, const float* __restrict__ d,
                               const float* __restrict__ g, const float* __restrict__ bt,
                               float* __restrict__ out) {
  int b = blockIdx.x, ch = threadIdx.x;
  float s = 0.f, s2 = 0.f;
  for (int i = 0; i < 96; ++i) {
    float v = x[(size_t)(b * 96 + i) * 256 + ch] + d[(size_t)(b * 96 + i) * 256 + ch];
    s += v;
    s2 += v * v;
  }
  float m = s * (1.f / 96.f);
  float var = s2 * (1.f / 96.f) - m * m;
  float rs = rsqrtf(var + EPS_);
  float gg = g[ch], bb = bt[ch];
  for (int i = 0; i < 96; ++i) {
    float v = x[(size_t)(b * 96 + i) * 256 + ch] + d[(size_t)(b * 96 + i) * 256 + ch];
    out[(size_t)(b * 96 + i) * 256 + ch] = (v - m) * rs * gg + bb;
  }
}

// ---------------------------------------------------------------- edge InstanceNorm: stats then normalize
__global__ void k_zero(float* __restrict__ p, int n) {
  int i = blockIdx.x * 256 + threadIdx.x;
  if (i < n) p[i] = 0.f;
}

__global__ void k_edge_stats(const float* __restrict__ x, float* __restrict__ stats) {
  int bx = blockIdx.x;                 // b*16 + chunk
  int b = bx >> 4, chunk = bx & 15;    // 16 chunks of 576 (i,j) pairs
  int t = threadIdx.x;
  int ch = t & 127, half = t >> 7;
  size_t ij0 = (size_t)chunk * 576 + (size_t)half * 288;
  float s = 0.f, s2 = 0.f;
  for (int q = 0; q < 288; ++q) {
    float v = x[((size_t)b * NN_ + ij0 + q) * 128 + ch];
    s += v;
    s2 += v * v;
  }
  __shared__ float ls[256], ls2[256];
  ls[t] = s;
  ls2[t] = s2;
  __syncthreads();
  if (half == 0) {
    s += ls[t + 128];
    s2 += ls2[t + 128];
    atomicAdd(&stats[(b * 128 + ch) * 2 + 0], s);
    atomicAdd(&stats[(b * 128 + ch) * 2 + 1], s2);
  }
}

__global__ void k_edge_norm(const float* __restrict__ x, const float* __restrict__ stats,
                            const float* __restrict__ g, const float* __restrict__ bt,
                            float* __restrict__ out, __bf16* __restrict__ obf) {
  size_t i = (size_t)blockIdx.x * 256 + threadIdx.x;
  if (i >= (size_t)R_ * 128) return;
  int ch = (int)(i & 127);
  int b = (int)((i >> 7) / NN_);
  float s = stats[(b * 128 + ch) * 2 + 0];
  float s2 = stats[(b * 128 + ch) * 2 + 1];
  float m = s * (1.f / (float)NN_);
  float var = s2 * (1.f / (float)NN_) - m * m;
  float rs = rsqrtf(var + EPS_);
  float y = (x[i] - m) * rs * g[ch] + bt[ch];
  out[i] = y;
  obf[i] = (__bf16)y;
}

// ---------------------------------------------------------------- output node heads
__global__ void k_out_nodes(const float* __restrict__ x, const float* __restrict__ tw1,
                            const float* __restrict__ tb1, const float* __restrict__ tw2,
                            const float* __restrict__ tb2, const float* __restrict__ pw1,
                            const float* __restrict__ pb1, const float* __restrict__ pw2,
                            const float* __restrict__ pb2, float* __restrict__ out) {
  __shared__ float xr[256], ht[256], hp[256];
  int r = blockIdx.x, t = threadIdx.x;
  xr[t] = x[r * 256 + t];
  __syncthreads();
  float a = tb1[t], c = pb1[t];
  for (int k = 0; k < 256; ++k) {
    a += xr[k] * tw1[t * 256 + k];
    c += xr[k] * pw1[t * 256 + k];
  }
  ht[t] = leaky_(a);
  hp[t] = leaky_(c);
  __syncthreads();
  if (t < 6) {
    float o = tb2[t];
    for (int k = 0; k < 256; ++k) o += ht[k] * tw2[t * 256 + k];
    out[r * 20 + t] = o;
  } else if (t < 20) {
    int u = t - 6;
    float o = pb2[u];
    for (int k = 0; k < 256; ++k) o += hp[k] * pw2[u * 256 + k];
    out[r * 20 + t] = o;
  }
}

// ---------------------------------------------------------------- out_e layer 2 (128->5)
__global__ void k_out_e2(const __bf16* __restrict__ h, const float* __restrict__ w,
                         const float* __restrict__ bias, float* __restrict__ out) {
  size_t idx = (size_t)blockIdx.x * 256 + threadIdx.x;
  if (idx >= (size_t)R_ * 5) return;
  size_t r = idx / 5;
  int c = (int)(idx % 5);
  const __bf16* hr = h + r * 128;
  float a = bias[c];
  for (int k = 0; k < 128; ++k) a += (float)hr[k] * w[c * 128 + k];
  out[idx] = a;
}

// ================================================================ host
template <int ACT, bool FILM, bool RES, bool OBF, bool OF32>
static void gemm(hipStream_t s, const __bf16* A, const __bf16* W, const float* bias,
                 const float* mul, const float* add, const float* res, __bf16* obf,
                 float* of32, int K, int Nout) {
  k_gemm<ACT, FILM, RES, OBF, OF32><<<dim3(R_ / 128, Nout / 64), dim3(256), 0, s>>>(
      A, W, bias, mul, add, res, obf, of32, K, Nout, NN_);
}

extern "C" void kernel_launch(void* const* d_in, const int* in_sizes, int n_in,
                              void* d_out, int out_size, void* d_ws, size_t ws_size,
                              hipStream_t stream) {
  (void)in_sizes; (void)n_in; (void)out_size; (void)ws_size;
  const float* nodes_in = (const float*)d_in[0];
  const float* edges_in = (const float*)d_in[1];
  const int*   tstep    = (const int*)d_in[2];
  auto F = [&](int i) { return (const float*)d_in[i]; };

  char* base = (char*)d_ws;
  size_t off = 0;
  auto alloc = [&](size_t bytes) -> void* {
    void* p = base + off;
    off = (off + bytes + 255) & ~(size_t)255;
    return p;
  };

  float*  nodes_cur = (float*)alloc((size_t)1536 * 256 * 4);
  float*  nodeA     = (float*)alloc((size_t)1536 * 256 * 4);
  float*  nodeB     = (float*)alloc((size_t)1536 * 256 * 4);
  float*  cond      = (float*)alloc(16 * 256 * 4);
  float*  fmul      = (float*)alloc(16 * 256 * 4);
  float*  faddv     = (float*)alloc(16 * 256 * 4);
  float*  stats     = (float*)alloc(16 * 128 * 2 * 4);
  float*  edges_cur = (float*)alloc((size_t)R_ * 128 * 4);
  float*  edges_tmp = (float*)alloc((size_t)R_ * 128 * 4);
  float*  wn        = (float*)alloc((size_t)R_ * 8 * 4);
  __bf16* gf        = (__bf16*)alloc((size_t)R_ * 640 * 2);
  __bf16* vbf       = gf;                       // reuse gf region after f-GEMM
  __bf16* wbf_act   = gf + (size_t)R_ * 256;
  __bf16* fbf       = (__bf16*)alloc((size_t)R_ * 256 * 2);
  __bf16* hbf       = (__bf16*)alloc((size_t)R_ * 256 * 2);
  __bf16* ebf       = (__bf16*)alloc((size_t)R_ * 128 * 2);

  // -- convert WMMA-path weights to bf16 (once per launch; deterministic)
  auto cvt = [&](int idx, int n) -> __bf16* {
    __bf16* p = (__bf16*)alloc((size_t)n * 2);
    k_f2bf<<<(n + 255) / 256, 256, 0, stream>>>(F(idx), p, n);
    return p;
  };
  __bf16 *Wf[4], *We[4], *Wv1[4], *Wv2[4], *Ww1[4], *Ww2[4], *Wme1[4], *Wme2[4];
  for (int l = 0; l < 4; ++l) {
    int b0 = 15 + l * 34;
    Wf[l]   = cvt(b0 + 0, 256 * 640);
    We[l]   = cvt(b0 + 6, 128 * 256);
    Wv1[l]  = cvt(b0 + 8, 256 * 256);
    Wv2[l]  = cvt(b0 + 10, 256 * 256);
    Ww1[l]  = cvt(b0 + 12, 256 * 256);
    Ww2[l]  = cvt(b0 + 14, 256 * 256);
    Wme1[l] = cvt(b0 + 26, 128 * 128);
    Wme2[l] = cvt(b0 + 28, 128 * 128);
  }
  __bf16* Wine2  = cvt(9, 128 * 128);
  __bf16* Woute1 = cvt(159, 128 * 128);

  // -- input embeddings
  k_in_n<<<1536, 256, 0, stream>>>(nodes_in, F(3), F(4), F(5), F(6), nodes_cur);
  k_in_e1<<<(int)(((size_t)R_ * 128 + 255) / 256), 256, 0, stream>>>(edges_in, F(7), F(8), hbf);
  gemm<1, false, false, false, true>(stream, hbf, Wine2, F(10), nullptr, nullptr, nullptr,
                                     nullptr, edges_cur, 128, 128);
  k_cond<<<16, 256, 0, stream>>>(tstep, F(11), F(12), F(13), F(14), cond);

  // -- transformer layers
  for (int l = 0; l < 4; ++l) {
    int b0 = 15 + l * 34;
    k_film<<<16, 256, 0, stream>>>(cond, F(b0 + 2), F(b0 + 3), F(b0 + 4), F(b0 + 5), fmul, faddv);
    k_pack_gf<<<R_, 256, 0, stream>>>(edges_cur, nodes_cur, gf);
    // f = leaky(FiLM(gf @ Wf^T + b))                -> bf16
    gemm<1, true, false, true, false>(stream, gf, Wf[l], F(b0 + 1), fmul, faddv, nullptr,
                                      fbf, nullptr, 640, 256);
    // edges_tmp = edges + leaky(f @ We^T + b)       -> f32
    gemm<1, false, true, false, true>(stream, fbf, We[l], F(b0 + 7), nullptr, nullptr,
                                      edges_cur, nullptr, edges_tmp, 256, 128);
    // v = (leaky(f@W1+b1)) @ W2 + b2
    gemm<1, false, false, true, false>(stream, fbf, Wv1[l], F(b0 + 9), nullptr, nullptr,
                                       nullptr, hbf, nullptr, 256, 256);
    gemm<0, false, false, true, false>(stream, hbf, Wv2[l], F(b0 + 11), nullptr, nullptr,
                                       nullptr, vbf, nullptr, 256, 256);
    // w likewise
    gemm<1, false, false, true, false>(stream, fbf, Ww1[l], F(b0 + 13), nullptr, nullptr,
                                       nullptr, hbf, nullptr, 256, 256);
    gemm<0, false, false, true, false>(stream, hbf, Ww2[l], F(b0 + 15), nullptr, nullptr,
                                       nullptr, wbf_act, nullptr, 256, 256);
    k_wn<<<R_ * 8 / 256, 256, 0, stream>>>(wbf_act, wn);
    k_attn<<<1536, 256, 0, stream>>>(wn, vbf, nodeA);
    k_lin256<<<1536, 256, 0, stream>>>(nodeA, F(b0 + 16), F(b0 + 17), nodeB);   // o-linear
    k_node_addnorm<<<16, 256, 0, stream>>>(nodes_cur, nodeB, F(b0 + 18), F(b0 + 19), nodes_cur);
    // n2 edge norm
    k_zero<<<16, 256, 0, stream>>>(stats, 4096);
    k_edge_stats<<<256, 256, 0, stream>>>(edges_tmp, stats);
    k_edge_norm<<<(int)(((size_t)R_ * 128 + 255) / 256), 256, 0, stream>>>(
        edges_tmp, stats, F(b0 + 20), F(b0 + 21), edges_cur, ebf);
    // node MLP + n3
    k_mlp256<<<1536, 256, 0, stream>>>(nodes_cur, F(b0 + 22), F(b0 + 23), F(b0 + 24),
                                       F(b0 + 25), nodeB);
    k_node_addnorm<<<16, 256, 0, stream>>>(nodes_cur, nodeB, F(b0 + 30), F(b0 + 31), nodes_cur);
    // edge MLP + residual + n4
    gemm<1, false, false, true, false>(stream, ebf, Wme1[l], F(b0 + 27), nullptr, nullptr,
                                       nullptr, hbf, nullptr, 128, 128);
    gemm<1, false, true, false, true>(stream, hbf, Wme2[l], F(b0 + 29), nullptr, nullptr,
                                      edges_cur, nullptr, edges_tmp, 128, 128);
    k_zero<<<16, 256, 0, stream>>>(stats, 4096);
    k_edge_stats<<<256, 256, 0, stream>>>(edges_tmp, stats);
    k_edge_norm<<<(int)(((size_t)R_ * 128 + 255) / 256), 256, 0, stream>>>(
        edges_tmp, stats, F(b0 + 32), F(b0 + 33), edges_cur, ebf);
  }

  // -- output heads
  float* out_f = (float*)d_out;
  k_out_nodes<<<1536, 256, 0, stream>>>(nodes_cur, F(151), F(152), F(153), F(154),
                                        F(155), F(156), F(157), F(158), out_f);
  gemm<1, false, false, true, false>(stream, ebf, Woute1, F(160), nullptr, nullptr, nullptr,
                                     hbf, nullptr, 128, 128);
  k_out_e2<<<(int)(((size_t)R_ * 5 + 255) / 256), 256, 0, stream>>>(
      hbf, F(161), F(162), out_f + 1536 * 20);
}
